// HNN_skip_68496138437415
// MI455X (gfx1250) — compile-verified, matrix operand order checked by reading, not run on hardware
//
#include <hip/hip_runtime.h>

typedef __attribute__((ext_vector_type(16))) _Float16 v16h;
typedef __attribute__((ext_vector_type(8)))  _Float16 v8h;
typedef __attribute__((ext_vector_type(8)))  float    v8f;
typedef __attribute__((ext_vector_type(4)))  float    v4f;
typedef __attribute__((ext_vector_type(4)))  int      v4i;

#define BB  8192
#define NL1 4096
#define NL2 2048
#define NL3 1024
#define NL4 512
#define NNZ1 262144
#define NNZ2 131072
#define NNZ3 65536
#define ND2 256
#define ND3 128

// ---------------- workspace layout (bytes) ----------------
#define OFF_W1F  0u                                   // 2048*4096*4 = 33554432
#define OFF_W2F  33554432u                            // 1024*2048*4 =  8388608
#define OFF_W3F  41943040u                            //  512*1024*4 =  2097152
#define OFF_W1H  44040192u                            // f16 weights  16777216
#define OFF_W2H  60817408u                            //               4194304
#define OFF_W3H  65011712u                            //               1048576
#define OFF_H1H  66060288u                            // 8192*2048*2 = 33554432
#define OFF_H2H  99614720u                            // 8192*1024*2 = 16777216
#define OFF_H3H  116391936u                           // 8192* 512*2 =  8388608
#define WF_TOTAL_F32 11010048

// ---------------- CDNA5 async-to-LDS support (guarded; falls back to VGPR copy) ----
#if __has_builtin(__builtin_amdgcn_global_load_async_to_lds_b128)
#define HAVE_ASYNC_LDS 1
#else
#define HAVE_ASYNC_LDS 0
#endif

template <typename T>
__device__ __forceinline__ T __attribute__((address_space(3)))* to_lds(T* p) {
  return (T __attribute__((address_space(3)))*)p;
}

// copy 16 bytes global -> LDS (async when available)
__device__ __forceinline__ void copy16_g2l(const _Float16* g, _Float16* l) {
#if HAVE_ASYNC_LDS
  __builtin_amdgcn_global_load_async_to_lds_b128((v4i*)g, to_lds((v4i*)l), 0, 0);
#else
  *(v8h*)l = *(const v8h*)g;
#endif
}

template <int N>
__device__ __forceinline__ void wait_asynccnt_le() {
#if HAVE_ASYNC_LDS
#if __has_builtin(__builtin_amdgcn_s_wait_asynccnt)
  __builtin_amdgcn_s_wait_asynccnt(N);
#else
  asm volatile("s_wait_asynccnt %0" ::"i"(N) : "memory");
#endif
#endif
}

// ---------------- helpers ----------------
__global__ __launch_bounds__(256) void zero_f32x4(float* __restrict__ p, int n4) {
  int i = blockIdx.x * blockDim.x + threadIdx.x;
  if (i < n4) { v4f z = {}; ((v4f*)p)[i] = z; }
}

__global__ __launch_bounds__(256) void scatter_add_edges(
    const int* __restrict__ in_idx, const int* __restrict__ out_idx,
    const float* __restrict__ w, float* __restrict__ Wd, int nnz, int ldk) {
  int e = blockIdx.x * blockDim.x + threadIdx.x;
  if (e < nnz) atomicAdd(Wd + (size_t)out_idx[e] * (size_t)ldk + (size_t)in_idx[e], w[e]);
}

__global__ __launch_bounds__(256) void cvt_f32_f16(const float* __restrict__ s,
                                                   _Float16* __restrict__ d, int n) {
  int i = blockIdx.x * blockDim.x + threadIdx.x;
  if (i < n) d[i] = (_Float16)s[i];
}

// ---------------- WMMA GEMM: Out = relu(A @ W^T + bias), f16 out ----------------
// Double-buffered LDS pipeline: tile t+1 streams in (async-to-LDS) while tile t computes.
// A: [M,K] (f32 or f16, row-major), W: [N,K] f16 row-major, Out: [M,N] f16.
// M,N multiples of 128; K multiple of 32 (holds for all three layers).
template <bool A_F32>
__global__ __launch_bounds__(256) void gemm_relu_wmma(
    const void* __restrict__ Aptr, const _Float16* __restrict__ W,
    const float* __restrict__ bias, _Float16* __restrict__ Out,
    int M, int N, int K)
{
  constexpr int MT = 128, NT = 128, KC = 32;
  constexpr int LDH = KC + 8;                       // LDS row stride in halves (80B)
  // per-thread async loads issued per tile (A_F32 stages A through VGPRs for the convert)
  constexpr int PER_TILE = A_F32 ? 2 : 4;
  __shared__ __align__(16) _Float16 As[2][MT * LDH];
  __shared__ __align__(16) _Float16 Ws[2][NT * LDH];

  const int tid    = threadIdx.x;
  const int lane   = tid & 31;
  const int wave   = tid >> 5;
  const int wm     = wave & 3;                      // 4 wave-rows of 32 output rows
  const int wn     = wave >> 2;                     // 2 wave-cols of 64 output cols
  const int lane16 = lane & 15;
  const int hsel   = lane >> 4;
  const int kb     = hsel * 8;                      // K sub-chunk base per 16-bit WMMA layout

  const int m_block = blockIdx.y * MT;
  const int n_block = blockIdx.x * NT;

  const int chunk = tid & 3;                        // which 8-half chunk within a 32-half row
  const int row0  = tid >> 2;                       // 0..63

  v8f acc[2][4];
#pragma unroll
  for (int i = 0; i < 2; ++i)
#pragma unroll
    for (int j = 0; j < 4; ++j) { v8f z = {}; acc[i][j] = z; }

  // stage tile t into LDS buffer `buf`
  auto issue_tile = [&](int t, int buf) {
    const int k0 = t * KC;
#pragma unroll
    for (int p = 0; p < 2; ++p) {
      const int row = row0 + 64 * p;
      // W operand: always async (f16 in memory already)
      copy16_g2l(W + (size_t)(n_block + row) * (size_t)K + (size_t)(k0 + chunk * 8),
                 &Ws[buf][row * LDH + chunk * 8]);
      // A operand
      if (A_F32) {
        const float* src = (const float*)Aptr +
                           (size_t)(m_block + row) * (size_t)K + (size_t)(k0 + chunk * 8);
        v4f f0 = *(const v4f*)src;
        v4f f1 = *(const v4f*)(src + 4);
        v8h av;
#pragma unroll
        for (int e = 0; e < 4; ++e) { av[e] = (_Float16)f0[e]; av[e + 4] = (_Float16)f1[e]; }
        *(v8h*)&As[buf][row * LDH + chunk * 8] = av;
      } else {
        copy16_g2l((const _Float16*)Aptr +
                       (size_t)(m_block + row) * (size_t)K + (size_t)(k0 + chunk * 8),
                   &As[buf][row * LDH + chunk * 8]);
      }
    }
  };

  const int T = K / KC;
  issue_tile(0, 0);
  if (T > 1) issue_tile(1, 1);

  for (int t = 0; t < T; ++t) {
    // our async loads for tile t must be in LDS; allow tile t+1's to stay in flight
    if (t + 1 < T) wait_asynccnt_le<PER_TILE>();
    else           wait_asynccnt_le<0>();
    __syncthreads();                                // all waves' tile-t data visible

    const int buf = t & 1;
    // Fragment loads per CDNA5 16-bit A/B layout:
    // lanes 0-15 hold K = 0..7,16..23 ; lanes 16-31 hold K = 8..15,24..31
    v16h afrag[2], bfrag[4];
#pragma unroll
    for (int i = 0; i < 2; ++i) {
      const _Float16* p = &As[buf][(wm * 32 + i * 16 + lane16) * LDH + kb];
      ((v8h*)&afrag[i])[0] = *(const v8h*)p;
      ((v8h*)&afrag[i])[1] = *(const v8h*)(p + 16);
    }
#pragma unroll
    for (int j = 0; j < 4; ++j) {
      const _Float16* p = &Ws[buf][(wn * 64 + j * 16 + lane16) * LDH + kb];
      ((v8h*)&bfrag[j])[0] = *(const v8h*)p;
      ((v8h*)&bfrag[j])[1] = *(const v8h*)(p + 16);
    }
#pragma unroll
    for (int i = 0; i < 2; ++i)
#pragma unroll
      for (int j = 0; j < 4; ++j)
        acc[i][j] = __builtin_amdgcn_wmma_f32_16x16x32_f16(
            false, afrag[i], false, bfrag[j], (short)0, acc[i][j], false, false);

    __syncthreads();                                // everyone done reading buf
    if (t + 2 < T) issue_tile(t + 2, buf);          // refill the freed buffer
  }

  // epilogue: bias + relu + f16 store. C layout: VGPR v -> M = 16i + v + 8*hsel, N = 16j + lane16.
#pragma unroll
  for (int j = 0; j < 4; ++j) {
    const int n  = n_block + wn * 64 + j * 16 + lane16;
    const float bv = bias[n];
#pragma unroll
    for (int i = 0; i < 2; ++i) {
      const int mbase = m_block + wm * 32 + i * 16 + hsel * 8;
#pragma unroll
      for (int v = 0; v < 8; ++v) {
        float r = acc[i][j][v] + bv;
        r = r > 0.f ? r : 0.f;
        Out[(size_t)(mbase + v) * (size_t)N + (size_t)n] = (_Float16)r;
      }
    }
  }
}

// ---------------- skip gathers + fc4 + readout, one wave32 per batch row ----------------
__global__ __launch_bounds__(256) void final_readout(
    const _Float16* __restrict__ h1, const _Float16* __restrict__ h2,
    const _Float16* __restrict__ h3,
    const int* __restrict__ d2, const float* __restrict__ wsk2, const float* __restrict__ bsk2,
    const int* __restrict__ d3, const float* __restrict__ wsk3, const float* __restrict__ bsk3,
    const float* __restrict__ W4, const float* __restrict__ b4,
    const float* __restrict__ Wro, const float* __restrict__ bro,
    float* __restrict__ out)
{
  const int row  = blockIdx.x * (blockDim.x >> 5) + (threadIdx.x >> 5);
  const int lane = threadIdx.x & 31;
  if (row >= BB) return;
  const _Float16* r1 = h1 + (size_t)row * NL2;
  const _Float16* r2 = h2 + (size_t)row * NL3;
  const _Float16* r3 = h3 + (size_t)row * NL4;

  float s2 = 0.f, s3 = 0.f, s4 = 0.f;
  for (int i = lane; i < ND2; i += 32) s2 += (float)r1[d2[i]] * wsk2[i];
  for (int i = lane; i < ND3; i += 32) s3 += (float)r2[d3[i]] * wsk3[i];
  for (int i = lane; i < NL4; i += 32) s4 += (float)r3[i] * W4[i];

#pragma unroll
  for (int off = 16; off > 0; off >>= 1) {
    s2 += __shfl_down(s2, off, 32);
    s3 += __shfl_down(s3, off, 32);
    s4 += __shfl_down(s4, off, 32);
  }
  if (lane == 0) {
    float sk2 = fmaxf(s2 + bsk2[0], 0.f);
    float sk3 = fmaxf(s3 + bsk3[0], 0.f);
    float f4  = fmaxf(s4 + b4[0],  0.f);
    out[row] = Wro[0] * sk2 + Wro[1] * sk3 + Wro[2] * f4 + bro[0];
  }
}

extern "C" void kernel_launch(void* const* d_in, const int* in_sizes, int n_in,
                              void* d_out, int out_size, void* d_ws, size_t ws_size,
                              hipStream_t stream) {
  (void)in_sizes; (void)n_in; (void)out_size; (void)ws_size;
  const float* x    = (const float*)d_in[0];
  const int*   in1  = (const int*)d_in[1];
  const int*   out1 = (const int*)d_in[2];
  const float* w1   = (const float*)d_in[3];
  const float* b1   = (const float*)d_in[4];
  const int*   in2  = (const int*)d_in[5];
  const int*   out2 = (const int*)d_in[6];
  const float* w2   = (const float*)d_in[7];
  const float* b2   = (const float*)d_in[8];
  const int*   d2   = (const int*)d_in[9];
  const float* wsk2 = (const float*)d_in[10];
  const float* bsk2 = (const float*)d_in[11];
  const int*   in3  = (const int*)d_in[12];
  const int*   out3 = (const int*)d_in[13];
  const float* w3   = (const float*)d_in[14];
  const float* b3   = (const float*)d_in[15];
  const int*   d3   = (const int*)d_in[16];
  const float* wsk3 = (const float*)d_in[17];
  const float* bsk3 = (const float*)d_in[18];
  const float* W4   = (const float*)d_in[19];
  const float* b4   = (const float*)d_in[20];
  const float* Wro  = (const float*)d_in[21];
  const float* bro  = (const float*)d_in[22];
  float* out = (float*)d_out;

  char* ws = (char*)d_ws;
  float*    W1f = (float*)(ws + OFF_W1F);
  float*    W2f = (float*)(ws + OFF_W2F);
  float*    W3f = (float*)(ws + OFF_W3F);
  _Float16* W1h = (_Float16*)(ws + OFF_W1H);
  _Float16* W2h = (_Float16*)(ws + OFF_W2H);
  _Float16* W3h = (_Float16*)(ws + OFF_W3H);
  _Float16* h1h = (_Float16*)(ws + OFF_H1H);
  _Float16* h2h = (_Float16*)(ws + OFF_H2H);
  _Float16* h3h = (_Float16*)(ws + OFF_H3H);

  // 1) zero dense f32 weights (deterministic every call)
  zero_f32x4<<<(WF_TOTAL_F32 / 4 + 255) / 256, 256, 0, stream>>>(W1f, WF_TOTAL_F32 / 4);

  // 2) scatter-add edges (duplicates accumulate, matching .at[].add)
  scatter_add_edges<<<NNZ1 / 256, 256, 0, stream>>>(in1, out1, w1, W1f, NNZ1, NL1);
  scatter_add_edges<<<NNZ2 / 256, 256, 0, stream>>>(in2, out2, w2, W2f, NNZ2, NL2);
  scatter_add_edges<<<NNZ3 / 256, 256, 0, stream>>>(in3, out3, w3, W3f, NNZ3, NL3);

  // 3) convert all three weight blocks f32 -> f16 in one pass (regions are contiguous)
  cvt_f32_f16<<<(WF_TOTAL_F32 + 255) / 256, 256, 0, stream>>>(W1f, W1h, WF_TOTAL_F32);

  // 4) WMMA GEMMs: h = relu(A @ W^T + b)
  gemm_relu_wmma<true ><<<dim3(NL2 / 128, BB / 128), 256, 0, stream>>>(
      (const void*)x,   W1h, b1, h1h, BB, NL2, NL1);
  gemm_relu_wmma<false><<<dim3(NL3 / 128, BB / 128), 256, 0, stream>>>(
      (const void*)h1h, W2h, b2, h2h, BB, NL3, NL2);
  gemm_relu_wmma<false><<<dim3(NL4 / 128, BB / 128), 256, 0, stream>>>(
      (const void*)h2h, W3h, b3, h3h, BB, NL4, NL3);

  // 5) skips + fc4 + readout
  final_readout<<<BB / 8, 256, 0, stream>>>(h1h, h2h, h3h,
                                            d2, wsk2, bsk2, d3, wsk3, bsk3,
                                            W4, b4, Wro, bro, out);
}
